// SingleStageDetector_17282948399270
// MI455X (gfx1250) — matrix operand order repaired and checked by reading, not compiled, not run on hardware
//
#include <hip/hip_runtime.h>
#include <hip/hip_bf16.h>
#include <math.h>

typedef __attribute__((ext_vector_type(16))) _Float16 v16h;
typedef __attribute__((ext_vector_type(8)))  float    v8f;

#define BDIM     64
#define IN_DIM   1280
#define HID      128
#define AC5      65          // 5*A + C_CLS
#define A_NUM    9
#define C_CLS    20
#define WF       28
#define SPATIAL  784         // 28*28
#define N_GT     40
#define M_POS    2048
#define NT       112         // output columns per block (7 wmma tiles)
#define KC       32          // K chunk per wmma
#define W1_STRIDE 40         // halfs; 80B row stride (16B multiple)
#define FT_STRIDE 40
#define HT_STRIDE 136        // halfs; 272B row stride (16B multiple)
#define W2_STRIDE 136

union HFrag { v16h v; float4 q[2]; };

// A-matrix 16x32 f16 fragment: lane holds row m; element e -> K = 2*hiA + e
// (contiguous, 2*hiA = 16 for lanes>=16), matching the B fragment.
__device__ inline v16h lds_load_a(const _Float16* rowp, int hiA) {
  HFrag f;
  f.q[0] = *reinterpret_cast<const float4*>(rowp + 2 * hiA);
  f.q[1] = *reinterpret_cast<const float4*>(rowp + 2 * hiA + 8);
  return f.v;
}
// B-matrix 32x16 f16 fragment: lane = column n; element e -> K = hiB + e
// (hiB = 16 for lanes>=16). Contiguous 16 halfs from a K-major LDS row.
__device__ inline v16h lds_load_b(const _Float16* rowp, int hiB) {
  HFrag f;
  f.q[0] = *reinterpret_cast<const float4*>(rowp + hiB);
  f.q[1] = *reinterpret_cast<const float4*>(rowp + hiB + 8);
  return f.v;
}

// Fused GEMM1 (W1 x feat) + bias + leaky-relu + GEMM2 (W2 x h) + bias -> x
// grid = 64 batches * 7 column tiles, 256 threads = 8 waves.
__global__ __launch_bounds__(256) void det_gemm_kernel(
    const float* __restrict__ F, const float* __restrict__ W1,
    const float* __restrict__ b1, const float* __restrict__ W2,
    const float* __restrict__ b2, float* __restrict__ xout) {
  __shared__ __align__(16) union {
    struct { _Float16 w1l[HID * W1_STRIDE]; _Float16 ftl[NT * FT_STRIDE]; } g1;
    _Float16 w2l[80 * W2_STRIDE];
  } u;
  __shared__ __align__(16) _Float16 htl[NT * HT_STRIDE];
  __shared__ float b1l[HID];
  __shared__ float b2l[80];

  const int tid  = threadIdx.x;
  const int lane = tid & 31;
  const int wave = tid >> 5;
  const int b    = blockIdx.x / 7;
  const int s0   = (blockIdx.x % 7) * NT;
  const int lr   = lane & 15;
  const int hiA  = (lane >= 16) ? 8 : 0;
  const int hiB  = (lane >= 16) ? 16 : 0;
  const int mhi  = (lane >= 16) ? 8 : 0;   // C/D row offset for upper half-wave

  if (tid < HID) b1l[tid] = b1[tid];
  if (tid < 80)  b2l[tid] = (tid < AC5) ? b2[tid] : 0.0f;

  v8f acc[7];
  #pragma unroll
  for (int n = 0; n < 7; ++n)
    #pragma unroll
    for (int r = 0; r < 8; ++r) acc[n][r] = 0.0f;

  // ---- GEMM1: h[128 x 112] = W1[128 x 1280] * feat[1280 x 112] ----
  for (int kc = 0; kc < IN_DIM / KC; ++kc) {
    const int k0 = kc * KC;
    __syncthreads();
    // stage W1 chunk (128 x 32) as f16, K-major rows
    for (int i = tid; i < HID * KC; i += 256) {
      int o = i >> 5, kk = i & 31;
      u.g1.w1l[o * W1_STRIDE + kk] = (_Float16)W1[o * IN_DIM + k0 + kk];
    }
    // stage feature chunk (32 x 112), transposed to [s][k] for B fragments
    for (int i = tid; i < KC * NT; i += 256) {
      int kk = i / NT, ss = i % NT;
      u.g1.ftl[ss * FT_STRIDE + kk] =
          (_Float16)F[((size_t)b * IN_DIM + (size_t)(k0 + kk)) * SPATIAL + s0 + ss];
    }
    __syncthreads();
    v16h af = lds_load_a(&u.g1.w1l[(wave * 16 + lr) * W1_STRIDE], hiA);
    #pragma unroll
    for (int n = 0; n < 7; ++n) {
      v16h bf = lds_load_b(&u.g1.ftl[(n * 16 + lr) * FT_STRIDE], hiB);
      acc[n] = __builtin_amdgcn_wmma_f32_16x16x32_f16(
          false, af, false, bf, (short)0, acc[n], false, false);
    }
  }
  __syncthreads();

  // bias + leaky relu, spill h to LDS transposed [s][o] for GEMM2 B fragments;
  // overlay region g1 is now dead -> load W2 tile (zero-padded to 80 rows)
  #pragma unroll
  for (int n = 0; n < 7; ++n) {
    #pragma unroll
    for (int r = 0; r < 8; ++r) {
      int m = wave * 16 + r + mhi;
      float h = acc[n][r] + b1l[m];
      h = (h > 0.0f) ? h : 0.01f * h;
      htl[(n * 16 + lr) * HT_STRIDE + m] = (_Float16)h;
    }
  }
  for (int i = tid; i < 80 * HID; i += 256) {
    int m = i >> 7, k = i & 127;
    float v = (m < AC5) ? W2[m * HID + k] : 0.0f;
    u.w2l[m * W2_STRIDE + k] = (_Float16)v;
  }
  __syncthreads();

  // ---- GEMM2: x[65 x 112] = W2[65 x 128] * h[128 x 112] ----
  for (int t = wave; t < 35; t += 8) {     // 5 M-tiles x 7 N-tiles
    const int mt = t / 7, nt = t % 7;
    v8f c;
    #pragma unroll
    for (int r = 0; r < 8; ++r) c[r] = 0.0f;
    #pragma unroll
    for (int kk = 0; kk < 4; ++kk) {
      v16h af = lds_load_a(&u.w2l[(mt * 16 + lr) * W2_STRIDE + kk * 32], hiA);
      v16h bf = lds_load_b(&htl[(nt * 16 + lr) * HT_STRIDE + kk * 32], hiB);
      c = __builtin_amdgcn_wmma_f32_16x16x32_f16(
          false, af, false, bf, (short)0, c, false, false);
    }
    #pragma unroll
    for (int r = 0; r < 8; ++r) {
      int m = mt * 16 + r + mhi;
      if (m < AC5) {
        int s = s0 + nt * 16 + lr;
        xout[((size_t)b * AC5 + m) * SPATIAL + s] = c[r] + b2l[m];
      }
    }
  }
}

// Per-anchor loss partials: pos (conf/reg/cls via IoU match) and neg (conf).
__global__ __launch_bounds__(256) void det_loss_kernel(
    const float* __restrict__ xbuf, const float* __restrict__ bboxes,
    const int* __restrict__ pos_idx, const int* __restrict__ neg_idx,
    const float* __restrict__ anc, float* __restrict__ confp,
    float* __restrict__ regp, float* __restrict__ clsp) {
  int t = blockIdx.x * blockDim.x + threadIdx.x;
  if (t >= 2 * M_POS) return;
  const bool is_pos = t < M_POS;
  int idx = is_pos ? pos_idx[t] : neg_idx[t - M_POS];
  int b   = idx / (A_NUM * SPATIAL);
  int rem = idx % (A_NUM * SPATIAL);
  int a   = rem / SPATIAL;
  int s   = rem % SPATIAL;
  const float* xb = xbuf + (size_t)b * AC5 * SPATIAL + s;

  float conf_logit = xb[(5 * a) * SPATIAL];
  float cs = 1.0f / (1.0f + expf(-conf_logit));
  if (!is_pos) { confp[t] = cs * cs; return; }  // gt_conf = 0
  confp[t] = (cs - 1.0f) * (cs - 1.0f);         // gt_conf = 1

  int hh = s / WF, ww = s % WF;
  float aw = anc[2 * a], ah = anc[2 * a + 1];
  float cx = ww + 0.5f, cy = hh + 0.5f;
  float px1 = cx - 0.5f * aw, py1 = cy - 0.5f * ah;
  float px2 = cx + 0.5f * aw, py2 = cy + 0.5f * ah;
  float area_p = aw * ah;

  const float* bb = bboxes + (size_t)b * N_GT * 5;
  float best = -1.0f; int bestn = 0;
  for (int n = 0; n < N_GT; ++n) {
    float bx1 = bb[n * 5 + 0], by1 = bb[n * 5 + 1];
    float bx2 = bb[n * 5 + 2], by2 = bb[n * 5 + 3];
    float area_b = (bx2 - bx1) * (by2 - by1);
    float iw = fminf(px2, bx2) - fmaxf(px1, bx1);
    float ih = fminf(py2, by2) - fmaxf(py1, by1);
    float inter = fmaxf(iw, 0.0f) * fmaxf(ih, 0.0f);
    float iou = inter / (area_p + area_b - inter);
    if (iou > best) { best = iou; bestn = n; }   // strict > keeps first max
  }
  float gx1 = bb[bestn * 5 + 0], gy1 = bb[bestn * 5 + 1];
  float gx2 = bb[bestn * 5 + 2], gy2 = bb[bestn * 5 + 3];
  int gtc = (int)bb[bestn * 5 + 4];
  gtc = gtc < 0 ? 0 : (gtc > C_CLS - 1 ? C_CLS - 1 : gtc);
  float g0 = 0.5f * (gx1 + gx2) - cx;
  float g1 = 0.5f * (gy1 + gy2) - cy;
  float g2 = logf((gx2 - gx1) / aw);
  float g3 = logf((gy2 - gy1) / ah);

  float o1 = xb[(5 * a + 1) * SPATIAL], o2 = xb[(5 * a + 2) * SPATIAL];
  float o3 = xb[(5 * a + 3) * SPATIAL], o4 = xb[(5 * a + 4) * SPATIAL];
  float f0 = 1.0f / (1.0f + expf(-o1)) - 0.5f;
  float f1 = 1.0f / (1.0f + expf(-o2)) - 0.5f;
  float d0 = f0 - g0, d1 = f1 - g1, d2 = o3 - g2, d3 = o4 - g3;
  regp[t] = d0 * d0 + d1 * d1 + d2 * d2 + d3 * d3;

  float cls[C_CLS]; float mx = -1e30f;
  #pragma unroll
  for (int j = 0; j < C_CLS; ++j) {
    cls[j] = xb[(5 * A_NUM + j) * SPATIAL];
    mx = fmaxf(mx, cls[j]);
  }
  float se = 0.0f;
  #pragma unroll
  for (int j = 0; j < C_CLS; ++j) se += expf(cls[j] - mx);
  clsp[t] = (mx + logf(se)) - cls[gtc];
}

// Deterministic single-block fixed-order reduction -> scalar loss.
__global__ __launch_bounds__(256) void det_reduce_kernel(
    const float* __restrict__ confp, const float* __restrict__ regp,
    const float* __restrict__ clsp, float* __restrict__ out) {
  __shared__ float sm[256];
  float acc = 0.0f;
  const float wc = 1.0f / (2.0f * M_POS);
  const float wr = 1.0f / M_POS;
  for (int i = threadIdx.x; i < 2 * M_POS; i += 256) acc += confp[i] * wc;
  for (int i = threadIdx.x; i < M_POS;     i += 256) acc += regp[i]  * wr;
  for (int i = threadIdx.x; i < M_POS;     i += 256) acc += clsp[i]  * wr;
  sm[threadIdx.x] = acc;
  __syncthreads();
  for (int st = 128; st > 0; st >>= 1) {
    if (threadIdx.x < st) sm[threadIdx.x] += sm[threadIdx.x + st];
    __syncthreads();
  }
  if (threadIdx.x == 0) out[0] = sm[0];
}

extern "C" void kernel_launch(void* const* d_in, const int* in_sizes, int n_in,
                              void* d_out, int out_size, void* d_ws, size_t ws_size,
                              hipStream_t stream) {
  (void)in_sizes; (void)n_in; (void)out_size; (void)ws_size;
  const float* F   = (const float*)d_in[0];
  const float* bbx = (const float*)d_in[1];
  const int*   pid = (const int*)d_in[2];
  const int*   nid = (const int*)d_in[3];
  const float* W1  = (const float*)d_in[4];
  const float* b1  = (const float*)d_in[5];
  const float* W2  = (const float*)d_in[6];
  const float* b2  = (const float*)d_in[7];
  const float* anc = (const float*)d_in[8];

  float* xbuf  = (float*)d_ws;                              // 64*65*784 floats
  float* confp = xbuf + (size_t)BDIM * AC5 * SPATIAL;       // 4096
  float* regp  = confp + 2 * M_POS;                         // 2048
  float* clsp  = regp + M_POS;                              // 2048

  det_gemm_kernel<<<dim3(BDIM * 7), dim3(256), 0, stream>>>(F, W1, b1, W2, b2, xbuf);
  det_loss_kernel<<<dim3((2 * M_POS + 255) / 256), dim3(256), 0, stream>>>(
      xbuf, bbx, pid, nid, anc, confp, regp, clsp);
  det_reduce_kernel<<<dim3(1), dim3(256), 0, stream>>>(confp, regp, clsp, (float*)d_out);
}
